// QueryPlanner_5944234738183
// MI455X (gfx1250) — matrix operand order repaired and checked by reading, not compiled, not run on hardware
//
#include <hip/hip_runtime.h>
#include <hip/hip_bf16.h>
#include <math.h>

typedef __attribute__((ext_vector_type(16))) _Float16 v16h;
typedef __attribute__((ext_vector_type(8)))  float    v8f;
typedef __attribute__((ext_vector_type(4)))  float    v4f;

namespace {
constexpr int   B       = 8;
constexpr int   N       = 1024;
constexpr int   P       = 64;
constexpr int   Dp      = 1024;
constexpr int   L       = N * P;
constexpr int   QMAX    = 32;
constexpr int   ANCHOR  = 4;
constexpr int   FLOWW   = 8;
constexpr int   TOPK    = 32;
constexpr float NEGV    = -10000.0f;
constexpr float LAMS    = 0.7f;
constexpr float LAML    = 0.3f;
}

// ---------------------------------------------------------------------------
// K1: exact f32 L2 norms of the 8 query vectors. One wave per query.
// ---------------------------------------------------------------------------
__global__ void qp_qnorm_kernel(const float* __restrict__ query_hidden,
                                float* __restrict__ norm_q) {
  int wid  = threadIdx.x >> 5;   // 8 waves, one per query
  int lane = threadIdx.x & 31;
  if (wid >= B) return;
  const float* src = query_hidden + (size_t)wid * Dp;
  float s = 0.0f;
  for (int i = lane * 4; i < Dp; i += 128) {
    float4 v = *reinterpret_cast<const float4*>(src + i);
    s += v.x * v.x + v.y * v.y + v.z * v.z + v.w * v.w;
  }
  #pragma unroll
  for (int d = 16; d > 0; d >>= 1) s += __shfl_down(s, d, 32);
  if (lane == 0) norm_q[wid] = sqrtf(s);
}

// ---------------------------------------------------------------------------
// K2: semantic scores via V_WMMA_F32_16X16X32_F16, fused with exact f32 page
// norms (single streaming pass over the 32 MB block_repr).
// One wave per (batch, 16-page tile). The batch's query vector is staged once
// per block into LDS, pre-arranged in per-k-step A-fragment order, so the hot
// loop holds a single 64-bit base address (B loads use immediate offsets) ->
// no register spills. All lanes read the same broadcast A fragment: that only
// replicates row M=0 into rows 1-15 of D, which are never read, so no
// predication / zero-fill is needed. Two WMMA accumulators break the RAW
// chain; sum-of-squares accumulates in v4f vector registers so it lowers to
// v_pk_fma_f32 on the loaded quads with zero packing movs. B-tile is streamed
// with non-temporal loads (read-once data).
// ---------------------------------------------------------------------------
__global__ void qp_sem_wmma_kernel(const float* __restrict__ block_repr,
                                   const float* __restrict__ query_hidden,
                                   const float* __restrict__ norm_q,
                                   float* __restrict__ scores_sem) {
  __shared__ __align__(32) _Float16 smemA[Dp];   // 2 KB: A fragments per k-step

  int w    = (blockIdx.x * blockDim.x + threadIdx.x) >> 5;
  int lane = threadIdx.x & 31;
  int b    = w >> 6;        // 64 tiles per batch; all 8 waves: same batch
  int tile = w & 63;

  // Stage query -> LDS in WMMA A-fragment order:
  //   step kk, halves [0..15]  = lane<16 flavor (K 0-15)
  //   step kk, halves [16..31] = lane>=16 flavor (K 16-31)
  for (int idx = threadIdx.x; idx < Dp; idx += 256) {
    int kk    = idx >> 5;
    int r     = idx & 31;
    int f16hi = (r >> 4) & 1;
    int pos   = r & 15;
    smemA[kk * 32 + f16hi * 16 + pos] =
        (_Float16)query_hidden[(size_t)b * Dp + idx];
  }
  __syncthreads();

  const float* kb = block_repr + ((size_t)b * N + (size_t)tile * 16) * Dp;
  int  col = lane & 15;     // B-matrix column (page within tile)
  bool hi  = lane >= 16;
  const float* kcol = kb + (size_t)col * Dp + (hi ? 16 : 0);
  const v16h* smA = reinterpret_cast<const v16h*>(smemA);

  v8f acc0 = {}, acc1 = {};
  v4f ssq0 = {}, ssq1 = {};   // vector sum-of-squares accumulators
  #pragma clang loop unroll_count(8)
  for (int kk = 0; kk < Dp / 32; ++kk) {
    v16h a = smA[kk * 2 + (hi ? 1 : 0)];   // 32-byte broadcast LDS read

    // 16 floats per lane in 4 float4 chunks: square + convert immediately so
    // the raw f32 values have short live ranges.
    v16h bm;
    const v4f* kp = reinterpret_cast<const v4f*>(kcol + kk * 32);
    #pragma unroll
    for (int c = 0; c < 4; ++c) {
      v4f x = __builtin_nontemporal_load(kp + c);
      if (c & 1) ssq1 += x * x; else ssq0 += x * x;   // v_pk_fma_f32, no movs
      bm[4 * c + 0] = (_Float16)x.x;
      bm[4 * c + 1] = (_Float16)x.y;
      bm[4 * c + 2] = (_Float16)x.z;
      bm[4 * c + 3] = (_Float16)x.w;
    }

    if (kk & 1)
      acc1 = __builtin_amdgcn_wmma_f32_16x16x32_f16(
          false, a, false, bm, (short)0, acc1, false, false);
    else
      acc0 = __builtin_amdgcn_wmma_f32_16x16x32_f16(
          false, a, false, bm, (short)0, acc0, false, false);
  }

  v4f sv = ssq0 + ssq1;                      // horizontal reduce once, at end
  float ssq = (sv.x + sv.y) + (sv.z + sv.w);
  float ssq_hi = __shfl_down(ssq, 16, 32);   // combine K 0-15 / K 16-31 halves
  if (!hi) {  // D row M=0 lives in acc[0], lanes 0-15 (N = lane)
    int page    = tile * 16 + col;
    float dot   = acc0[0] + acc1[0];
    float nk    = sqrtf(ssq + ssq_hi);
    float denom = fmaxf(nk, 1e-12f) * fmaxf(norm_q[b], 1e-12f);
    scores_sem[b * N + page] = dot / denom;
  }
}

// ---------------------------------------------------------------------------
// K3: lexical scores + per-page "any valid" flag.
// Block = 8 pages (1 wave each). Query ids cached in LDS. Each wave stages
// its page's 64 gather indices into LDS via the gfx1250 async-to-LDS path
// (global_load_async_to_lds_b64, tracked by ASYNCcnt).
// ---------------------------------------------------------------------------
__global__ void qp_lex_kernel(const int* __restrict__ input_ids,
                              const float* __restrict__ tok_w,
                              const int* __restrict__ q_start,
                              const int* __restrict__ q_end,
                              const unsigned char* __restrict__ tok_valid,
                              const unsigned char* __restrict__ page_valid,
                              const int* __restrict__ page_indices,
                              float* __restrict__ scores_lex,
                              int* __restrict__ valid_any) {
  __shared__ int qids[QMAX];
  __shared__ int s_idx[8 * P];   // 8 pages x 64 gather indices
  int b     = blockIdx.x >> 7;   // N/8 = 128 blocks per batch
  int chunk = blockIdx.x & 127;
  int wave  = threadIdx.x >> 5;
  int lane  = threadIdx.x & 31;
  int n     = chunk * 8 + wave;
  size_t pbase = ((size_t)b * N + n) * P;

  // Async-stage this wave's 64 page_indices (2 ints per lane, b64 each).
  {
    const int* gsrc    = page_indices + pbase + lane * 2;
    unsigned   lds_off = (unsigned)(uintptr_t)&s_idx[wave * P + lane * 2];
    asm volatile("global_load_async_to_lds_b64 %0, %1, off"
                 :: "v"(lds_off), "v"(gsrc) : "memory");
  }

  if (threadIdx.x < QMAX) {
    int j  = threadIdx.x;
    int s0 = q_start[b], e0 = q_end[b];
    int qs = min(max(s0, 0), L - 1);
    int qe = min(max(e0, 0), L - 1);
    int pos = min(qs + j, L - 1);
    bool m = (j <= qe - qs) && (e0 >= s0) &&
             (tok_valid[(size_t)b * L + pos] != 0);
    qids[j] = m ? input_ids[(size_t)b * L + pos] : -1;  // -1 never matches
  }

  asm volatile("s_wait_asynccnt 0" ::: "memory");  // per-wave staging done
  __syncthreads();                                 // qids visible block-wide

  float s  = 0.0f;
  int   av = 0;
  #pragma unroll
  for (int t = 0; t < 2; ++t) {
    int p  = lane + t * 32;
    int pv = page_valid[pbase + p];
    av |= pv;
    int idx = s_idx[wave * P + p];
    int tok = input_ids[(size_t)b * L + idx];
    bool match = false;
    #pragma unroll
    for (int u = 0; u < QMAX; ++u) match = match || (tok == qids[u]);
    if (pv && match) s += tok_w[(size_t)b * L + idx];
  }
  #pragma unroll
  for (int d = 16; d > 0; d >>= 1) {
    s  += __shfl_down(s, d, 32);
    av |= __shfl_down(av, d, 32);
  }
  if (lane == 0) {
    scores_lex[b * N + n] = s;
    valid_any[b * N + n]  = av;
  }
}

// ---------------------------------------------------------------------------
// K4: masks, min-max normalization, mixing, anchor/flow/flash top-k, output.
// One block of 1024 threads per batch (n = threadIdx.x).
// ---------------------------------------------------------------------------
__device__ inline float qp_block_max(float v, float* buf, int n) {
  buf[n] = v; __syncthreads();
  for (int off = 512; off > 0; off >>= 1) {
    if (n < off) buf[n] = fmaxf(buf[n], buf[n + off]);
    __syncthreads();
  }
  float r = buf[0]; __syncthreads();
  return r;
}
__device__ inline float qp_block_min(float v, float* buf, int n) {
  buf[n] = v; __syncthreads();
  for (int off = 512; off > 0; off >>= 1) {
    if (n < off) buf[n] = fminf(buf[n], buf[n + off]);
    __syncthreads();
  }
  float r = buf[0]; __syncthreads();
  return r;
}

__global__ void __launch_bounds__(1024)
qp_final_kernel(const int* __restrict__ query_pos,
                const int* __restrict__ token2page,
                const int* __restrict__ segment_ids,
                const float* __restrict__ scores_sem,
                const float* __restrict__ scores_lex,
                const int* __restrict__ valid_any,
                float* __restrict__ out) {
  __shared__ float rv[N];
  __shared__ int   ri[N];
  __shared__ int   s_qp;

  int b = blockIdx.x;
  int n = threadIdx.x;

  if (n == 0) {
    int qp = min(max(query_pos[b], 0), L - 1);
    s_qp = max(token2page[(size_t)b * L + qp], 0);
  }
  __syncthreads();
  int qpage = s_qp;

  int  seg   = segment_ids[b * N + n];
  bool valid = (valid_any[b * N + n] != 0) && (seg >= 0) && (n <= qpage);
  float sem  = scores_sem[b * N + n];
  float lex  = scores_lex[b * N + n];

  float inf = __builtin_inff();
  float smin = qp_block_min(valid ? sem :  inf, rv, n);
  float smax = qp_block_max(valid ? sem : -inf, rv, n);
  float lmin = qp_block_min(valid ? lex :  inf, rv, n);
  float lmax = qp_block_max(valid ? lex : -inf, rv, n);

  float srng = smax - smin;
  float lrng = lmax - lmin;
  float sem_n = (!valid || !(srng >= 1e-6f)) ? 0.0f : (sem - smin) / srng;
  float lex_n = (!valid || !(lrng >= 1e-6f)) ? 0.0f : (lex - lmin) / lrng;
  float mix   = (LAMS * sem_n + LAML * lex_n) / (LAMS + LAML);
  float fin   = valid ? mix : NEGV;

  // anchor = seg0 & valid & cumsum(is_seg0) <= ANCHOR  (inclusive scan)
  bool is0 = (seg == 0) && valid;
  ri[n] = is0 ? 1 : 0; __syncthreads();
  for (int off = 1; off < N; off <<= 1) {
    int add = (n >= off) ? ri[n - off] : 0;
    __syncthreads();
    ri[n] += add;
    __syncthreads();
  }
  int csum = ri[n]; __syncthreads();
  bool anchor = is0 && (csum <= ANCHOR);

  int  lower = max(qpage - FLOWW, 0);
  bool flow  = ((n >= lower) && (n <= qpage) && valid) || (n == qpage);
  bool base_keep = anchor || flow;
  bool cand  = valid && !base_keep;
  float scand = cand ? fin : NEGV;

  // iterative top-32 argmax (smaller index wins ties, matching lax.top_k)
  bool flash = false;
  float cur = scand;
  for (int r = 0; r < TOPK; ++r) {
    rv[n] = cur; ri[n] = n; __syncthreads();
    for (int off = 512; off > 0; off >>= 1) {
      if (n < off) {
        float v2 = rv[n + off]; int i2 = ri[n + off];
        if (v2 > rv[n] || (v2 == rv[n] && i2 < ri[n])) { rv[n] = v2; ri[n] = i2; }
      }
      __syncthreads();
    }
    int sel = ri[0]; __syncthreads();
    if (n == sel) { flash = flash || cand; cur = -__builtin_inff(); }
    __syncthreads();
  }

  out[b * N + n] = (base_keep || flash) ? 1.0f : 0.0f;
}

// ---------------------------------------------------------------------------
extern "C" void kernel_launch(void* const* d_in, const int* in_sizes, int n_in,
                              void* d_out, int out_size, void* d_ws, size_t ws_size,
                              hipStream_t stream) {
  (void)in_sizes; (void)n_in; (void)out_size; (void)ws_size;
  const float*         block_repr   = (const float*)d_in[0];
  const float*         query_hidden = (const float*)d_in[1];
  const int*           query_pos    = (const int*)d_in[2];
  const int*           input_ids    = (const int*)d_in[3];
  const float*         tok_w        = (const float*)d_in[4];
  const int*           q_start      = (const int*)d_in[5];
  const int*           q_end        = (const int*)d_in[6];
  const int*           segment_ids  = (const int*)d_in[7];
  const unsigned char* page_valid   = (const unsigned char*)d_in[8];
  const int*           page_indices = (const int*)d_in[9];
  const int*           token2page   = (const int*)d_in[10];
  const unsigned char* token_valid  = (const unsigned char*)d_in[11];

  float* ws         = (float*)d_ws;
  float* norm_q     = ws;                          // B
  float* scores_sem = ws + B;                      // B*N
  float* scores_lex = scores_sem + B * N;          // B*N
  int*   valid_any  = (int*)(scores_lex + B * N);  // B*N
  float* out        = (float*)d_out;

  // K1: 8 query norms, one wave each
  qp_qnorm_kernel<<<1, 256, 0, stream>>>(query_hidden, norm_q);

  // K2: 512 waves (8 batches x 64 tiles), 8 waves per block; fused norms
  qp_sem_wmma_kernel<<<(B * (N / 16)) / 8, 256, 0, stream>>>(
      block_repr, query_hidden, norm_q, scores_sem);

  // K3: 1024 blocks (8 pages each), async-to-LDS index staging
  qp_lex_kernel<<<B * (N / 8), 256, 0, stream>>>(
      input_ids, tok_w, q_start, q_end, token_valid, page_valid,
      page_indices, scores_lex, valid_any);

  // K4: one 1024-thread block per batch
  qp_final_kernel<<<B, 1024, 0, stream>>>(
      query_pos, token2page, segment_ids, scores_sem, scores_lex,
      valid_any, out);
}
